// ContactHybridMamba_46342697123957
// MI455X (gfx1250) — matrix operand order repaired and checked by reading, not compiled, not run on hardware
//
#include <hip/hip_runtime.h>
#include <math.h>

// ---------------- model dims ----------------
#define BB 2
#define NN 2048
#define DIMD 256
#define DINNER 512
#define DSTATE 16
#define DTRANK 16
#define XDBLW 48       // DT_RANK + 2*D_STATE
#define CONDD 768
#define NLAT 128
#define NHEAD 8
#define HDIM 32
#define INCH 523       // 8 + 512 + 3
#define BN (BB*NN)     // 4096

typedef __attribute__((ext_vector_type(16))) _Float16 v16h;
typedef __attribute__((ext_vector_type(8)))  _Float16 v8h;
typedef __attribute__((ext_vector_type(8)))  float    v8f;
typedef __attribute__((ext_vector_type(4)))  float    v4f;

// ---------------- device math helpers ----------------
__device__ __forceinline__ float gelu_f(float x)     { return 0.5f * x * (1.0f + erff(x * 0.70710678118654752f)); }
__device__ __forceinline__ float silu_f(float x)     { return x / (1.0f + expf(-x)); }
__device__ __forceinline__ float softplus_f(float x) { return (x > 20.0f) ? x : log1pf(expf(x)); }

// =====================================================================
// WMMA GEMM: C[M,N] = act(A[M,K](lda) @ W[K,N] + bias)
// Requirements (guaranteed by all call sites): M % 128 == 0, N % 16 == 0.
// Block = 256 threads = 8 waves; block tile = 128(M) x 16(N); K chunks of 32.
// A/B staged through LDS as f16 (B reused by all 8 waves); fragments read as
// contiguous LDS vector loads per CDNA5 ISA 7.12.2 layouts; guarded loads only
// in the final ragged-K chunk so the hot path has no per-lane predication.
// =====================================================================
__global__ void k_gemm(const float* __restrict__ A, const float* __restrict__ W,
                       const float* __restrict__ bias, float* __restrict__ C,
                       int M, int Nc, int K, int lda, int act)
{
    __shared__ __align__(32) _Float16 sA[128][40];  // 32 k + 8 pad halfs
    __shared__ __align__(32) _Float16 sB[16][64];   // [n][k], padded row

    int tid  = threadIdx.x;
    int lane = tid & 31;
    int wave = tid >> 5;
    int n0   = blockIdx.x * 16;
    int m0b  = blockIdx.y * 128;

    // staging coordinates
    int ar = tid >> 1;            // A row 0..127
    int ak = (tid & 1) * 16;      // A k offset 0 or 16
    int bk = tid >> 3;            // B k 0..31
    int bn = (tid & 7) * 2;       // B n 0..14

    // fragment coordinates
    int ml = wave * 16 + (lane & 15);
    int nl = lane & 15;
    int hi = lane >> 4;

    const bool a_aligned = ((lda & 3) == 0);
    const float* arow = A + (size_t)(m0b + ar) * lda + ak;

    v8f acc = {};
    for (int k0 = 0; k0 < K; k0 += 32) {
        // ---- stage A (128x32) and B (32x16, transposed) into LDS as f16 ----
        if (k0 + 32 <= K) {
            if (a_aligned) {
#pragma unroll
                for (int i = 0; i < 4; ++i) {
                    v4f f = *(const v4f*)(arow + k0 + 4 * i);
#pragma unroll
                    for (int j = 0; j < 4; ++j) sA[ar][ak + 4 * i + j] = (_Float16)f[j];
                }
            } else {
#pragma unroll
                for (int i = 0; i < 16; ++i) sA[ar][ak + i] = (_Float16)arow[k0 + i];
            }
            const float* wrow = W + (size_t)(k0 + bk) * Nc + n0 + bn;
            sB[bn][bk]     = (_Float16)wrow[0];
            sB[bn + 1][bk] = (_Float16)wrow[1];
        } else {
#pragma unroll
            for (int i = 0; i < 16; ++i) {
                int kk = k0 + ak + i;
                sA[ar][ak + i] = (_Float16)((kk < K) ? arow[k0 + i] : 0.0f);
            }
            float w0 = 0.0f, w1 = 0.0f;
            if (k0 + bk < K) {
                const float* wrow = W + (size_t)(k0 + bk) * Nc + n0 + bn;
                w0 = wrow[0]; w1 = wrow[1];
            }
            sB[bn][bk]     = (_Float16)w0;
            sB[bn + 1][bk] = (_Float16)w1;
        }
        __syncthreads();

        // ---- fragments from LDS (contiguous vector reads) ----
        v8h a_lo = *(const v8h*)&sA[ml][hi * 8];        // k = hi*8 .. hi*8+7
        v8h a_hi = *(const v8h*)&sA[ml][16 + hi * 8];   // k = 16+hi*8 .. +7
        v16h af;
#pragma unroll
        for (int i = 0; i < 8; ++i) { af[i] = a_lo[i]; af[8 + i] = a_hi[i]; }
        v8h b_lo = *(const v8h*)&sB[nl][hi * 8];         // k = hi*8 .. hi*8+7
        v8h b_hi = *(const v8h*)&sB[nl][16 + hi * 8];    // k = 16+hi*8 .. +7
        v16h bf;
#pragma unroll
        for (int i = 0; i < 8; ++i) { bf[i] = b_lo[i]; bf[8 + i] = b_hi[i]; }

        acc = __builtin_amdgcn_wmma_f32_16x16x32_f16(false, af, false, bf,
                                                     (short)0, acc, false, false);
        __syncthreads();
    }

    int ncol = n0 + nl;
    float bv = bias ? bias[ncol] : 0.0f;
#pragma unroll
    for (int v = 0; v < 8; ++v) {
        int mr = m0b + wave * 16 + v + hi * 8;          // f32 D: V0..7 -> M=v (+8 hi)
        float o = acc[v] + bv;
        if (act == 1) o = gelu_f(o);
        else if (act == 2) o = softplus_f(o);
        C[(size_t)mr * Nc + ncol] = o;
    }
}

// ---------------- layernorm over 256 channels (one wave per row) ----------------
__global__ void k_ln256(const float* __restrict__ X, const float* __restrict__ g,
                        const float* __restrict__ b, float* __restrict__ Y, int rows)
{
    int wave = threadIdx.x >> 5, lane = threadIdx.x & 31;
    int r = blockIdx.x * 8 + wave;
    if (r >= rows) return;
    const float* x = X + (size_t)r * DIMD;
    float v[8]; float s = 0.0f;
#pragma unroll
    for (int i = 0; i < 8; ++i) { v[i] = x[lane + 32 * i]; s += v[i]; }
    for (int o = 16; o; o >>= 1) s += __shfl_xor(s, o, 32);
    float mean = s * (1.0f / 256.0f);
    float q = 0.0f;
#pragma unroll
    for (int i = 0; i < 8; ++i) { float d = v[i] - mean; q += d * d; }
    for (int o = 16; o; o >>= 1) q += __shfl_xor(q, o, 32);
    float inv = rsqrtf(q * (1.0f / 256.0f) + 1e-5f);
    float* y = Y + (size_t)r * DIMD;
#pragma unroll
    for (int i = 0; i < 8; ++i) {
        int c = lane + 32 * i;
        y[c] = (v[i] - mean) * inv * g[c] + b[c];
    }
}

// ---------------- small linear (cond / FiLM paths; grid.y = batch) ----------------
__global__ void k_slin(const float* __restrict__ in, const float* __restrict__ w,
                       const float* __restrict__ bias, float* __restrict__ out,
                       int K, int Nout, int act)
{
    int j = blockIdx.x * blockDim.x + threadIdx.x;
    int b = blockIdx.y;
    if (j >= Nout) return;
    float s = bias[j];
    for (int k = 0; k < K; ++k) s += in[b * K + k] * w[k * Nout + j];
    if (act == 1) s = gelu_f(s);
    out[b * Nout + j] = s;
}

// ---------------- feature concat [x | point_feat | xyz] ----------------
__global__ void k_feat(const float* __restrict__ x, const float* __restrict__ pf,
                       const float* __restrict__ xyz, float* __restrict__ feat)
{
    int id = blockIdx.x * blockDim.x + threadIdx.x;
    if (id >= BN * INCH) return;
    int c = id % INCH; int r = id / INCH;
    float v;
    if (c < 8)        v = x[(size_t)r * 8 + c];
    else if (c < 520) v = pf[(size_t)r * 512 + (c - 8)];
    else              v = xyz[(size_t)r * 3 + (c - 520)];
    feat[id] = v;
}

// ---------------- per-batch xyz min/max ----------------
__global__ void k_minmax(const float* __restrict__ xyz, float* __restrict__ mnmx)
{
    __shared__ float smn[3][256], smx[3][256];
    int b = blockIdx.x, t = threadIdx.x;
    float mn[3] = {1e30f, 1e30f, 1e30f}, mx[3] = {-1e30f, -1e30f, -1e30f};
    for (int i = t; i < NN; i += 256)
        for (int d = 0; d < 3; ++d) {
            float v = xyz[((size_t)(b * NN + i)) * 3 + d];
            mn[d] = fminf(mn[d], v); mx[d] = fmaxf(mx[d], v);
        }
    for (int d = 0; d < 3; ++d) { smn[d][t] = mn[d]; smx[d][t] = mx[d]; }
    __syncthreads();
    for (int o = 128; o; o >>= 1) {
        if (t < o)
            for (int d = 0; d < 3; ++d) {
                smn[d][t] = fminf(smn[d][t], smn[d][t + o]);
                smx[d][t] = fmaxf(smx[d][t], smx[d][t + o]);
            }
        __syncthreads();
    }
    if (t < 3) { mnmx[b * 6 + t] = smn[t][0]; mnmx[b * 6 + 3 + t] = smx[t][0]; }
}

// ---------------- Hilbert / Morton codes ----------------
__device__ unsigned long long hilbert3(unsigned long long* X, int nbits)
{
    unsigned long long M = 1ull << (nbits - 1);
    for (unsigned long long Q = M; Q > 1; Q >>= 1) {
        unsigned long long P = Q - 1;
        for (int i = 0; i < 3; ++i) {
            bool c = (X[i] & Q) != 0;
            unsigned long long t = (X[0] ^ X[i]) & P;
            unsigned long long nX0 = c ? (X[0] ^ P) : (X[0] ^ t);
            unsigned long long nXi = c ? X[i] : (X[i] ^ t);
            X[0] = nX0;
            if (i > 0) X[i] = nXi;
        }
    }
    X[1] ^= X[0]; X[2] ^= X[1];
    unsigned long long t = 0;
    for (unsigned long long Q = M; Q > 1; Q >>= 1)
        if (X[2] & Q) t ^= (Q - 1);
    for (int i = 0; i < 3; ++i) X[i] ^= t;
    unsigned long long h = 0;
    for (int bb = nbits - 1; bb >= 0; --bb)
        for (int d = 0; d < 3; ++d)
            h = (h << 1) | ((X[d] >> bb) & 1ull);
    return h;
}

__device__ __forceinline__ unsigned long long spread10(unsigned long long v)
{
    v &= 1023ull;
    v = (v | (v << 16)) & 50331903ull;
    v = (v | (v << 8))  & 50393103ull;
    v = (v | (v << 4))  & 51130563ull;
    v = (v | (v << 2))  & 153391689ull;
    return v;
}

__global__ void k_codes(const float* __restrict__ xyz, const float* __restrict__ mnmx,
                        unsigned long long* __restrict__ code, int* __restrict__ idx, int mode)
{
    int id = blockIdx.x * blockDim.x + threadIdx.x;
    if (id >= BN) return;
    int b = id / NN, i = id % NN;
    unsigned long long q[3];
    for (int d = 0; d < 3; ++d) {
        float mn = mnmx[b * 6 + d], mx = mnmx[b * 6 + 3 + d];
        float nrm = (xyz[(size_t)id * 3 + d] - mn) / (mx - mn + 1e-8f);
        if (mode == 0) { nrm *= 0.999f; q[d] = (unsigned long long)(nrm * 65535.0f); }
        else           { q[d] = (unsigned long long)(nrm * 1023.0f); }
    }
    unsigned long long c;
    if (mode == 0) c = hilbert3(q, 16);
    else           c = spread10(q[0]) | (spread10(q[1]) << 1) | (spread10(q[2]) << 2);
    code[id] = c;
    idx[id] = i;
}

// ---------------- in-LDS bitonic argsort (stable tie-break), one block per batch ----------------
__global__ void k_bitonic(unsigned long long* __restrict__ codes, int* __restrict__ idx)
{
    __shared__ unsigned long long sk[NN];
    __shared__ int sidx[NN];
    int b = blockIdx.x;
    for (int i = threadIdx.x; i < NN; i += blockDim.x) { sk[i] = codes[b * NN + i]; sidx[i] = idx[b * NN + i]; }
    __syncthreads();
    for (int k = 2; k <= NN; k <<= 1) {
        for (int j = k >> 1; j > 0; j >>= 1) {
            for (int i = threadIdx.x; i < NN; i += blockDim.x) {
                int p = i ^ j;
                if (p > i) {
                    bool up = ((i & k) == 0);
                    unsigned long long a = sk[i], c = sk[p];
                    int ia = sidx[i], ic = sidx[p];
                    bool sw = up ? (a > c || (a == c && ia > ic))
                                 : (a < c || (a == c && ia < ic));
                    if (sw) { sk[i] = c; sk[p] = a; sidx[i] = ic; sidx[p] = ia; }
                }
            }
            __syncthreads();
        }
    }
    for (int i = threadIdx.x; i < NN; i += blockDim.x) { codes[b * NN + i] = sk[i]; idx[b * NN + i] = sidx[i]; }
}

__global__ void k_invperm(const int* __restrict__ si, int* __restrict__ ui)
{
    int id = blockIdx.x * blockDim.x + threadIdx.x;
    if (id >= BN) return;
    ui[(id / NN) * NN + si[id]] = id % NN;
}

// ---------------- row gather by permutation ----------------
__global__ void k_gather(const float* __restrict__ in, const int* __restrict__ perm,
                         float* __restrict__ out, int C)
{
    int id = blockIdx.x * blockDim.x + threadIdx.x;
    if (id >= BN * C) return;
    int c = id % C; int j = (id / C) % NN; int b = id / (C * NN);
    out[id] = in[((size_t)(b * NN + perm[b * NN + j])) * C + c];
}

// ---------------- causal depthwise conv1d + SiLU (dir=1 means reversed seq) ----------------
__global__ void k_conv(const float* __restrict__ uz, const float* __restrict__ cw,
                       const float* __restrict__ cb, float* __restrict__ uc, int dir)
{
    int id = blockIdx.x * blockDim.x + threadIdx.x;
    if (id >= BN * DINNER) return;
    int c = id % DINNER; int t = (id / DINNER) % NN; int b = id / (DINNER * NN);
    float s = cb[c];
#pragma unroll
    for (int j = 0; j < 4; ++j) {
        int tt = dir ? (t + 3 - j) : (t - 3 + j);
        if (tt >= 0 && tt < NN)
            s += cw[c * 4 + j] * uz[((size_t)(b * NN + tt)) * 1024 + c];
    }
    uc[id] = silu_f(s);
}

// ---------------- selective scan: one thread per (b, d_inner) channel ----------------
__global__ void k_scan(const float* __restrict__ dt, const float* __restrict__ uc,
                       const float* __restrict__ xdbl, const float* __restrict__ A_log,
                       const float* __restrict__ Dp, const float* __restrict__ uz,
                       float* __restrict__ y, int dir)
{
    int id = blockIdx.x * blockDim.x + threadIdx.x;
    if (id >= BB * DINNER) return;
    int d = id % DINNER; int b = id / DINNER;
    float A[DSTATE], h[DSTATE];
#pragma unroll
    for (int s = 0; s < DSTATE; ++s) { A[s] = -expf(A_log[d * DSTATE + s]); h[s] = 0.0f; }
    float Dd = Dp[d];
    for (int step = 0; step < NN; ++step) {
        int t = dir ? (NN - 1 - step) : step;
        size_t row = (size_t)(b * NN + t);
        float dtv = dt[row * DINNER + d];
        float uv  = uc[row * DINNER + d];
        float du  = dtv * uv;
        const float* xd = xdbl + row * XDBLW;
        float yv = 0.0f;
#pragma unroll
        for (int s = 0; s < DSTATE; ++s) {
            float dA = expf(dtv * A[s]);
            h[s] = h[s] * dA + du * xd[16 + s];
            yv += h[s] * xd[32 + s];
        }
        float zv = uz[row * 1024 + 512 + d];
        y[row * DINNER + d] = (yv + uv * Dd) * silu_f(zv);
    }
}

// ---------------- FiLM + residual accumulate ----------------
__global__ void k_film(float* __restrict__ xs, const float* __restrict__ mout,
                       const float* __restrict__ gmm, const float* __restrict__ bet)
{
    int id = blockIdx.x * blockDim.x + threadIdx.x;
    if (id >= BN * DIMD) return;
    int c = id % DIMD; int b = id / (NN * DIMD);
    xs[id] += gmm[b * DIMD + c] * mout[id] + bet[b * DIMD + c];
}

// ---------------- concat two [rows,C] buffers ----------------
__global__ void k_cat2(const float* __restrict__ a, const float* __restrict__ b2,
                       float* __restrict__ out, int C1, int C2)
{
    int C = C1 + C2;
    int id = blockIdx.x * blockDim.x + threadIdx.x;
    if (id >= BN * C) return;
    int c = id % C; int r = id / C;
    out[id] = (c < C1) ? a[(size_t)r * C1 + c] : b2[(size_t)r * C2 + (c - C1)];
}

// ---------------- attention: one block per (b,h,q); LDS softmax ----------------
__global__ void k_attn(const float* __restrict__ Q, const float* __restrict__ Kp,
                       const float* __restrict__ Vp, float* __restrict__ O,
                       int Lq, int Lk)
{
    extern __shared__ float sc[];      // Lk scores
    __shared__ float red[128];
    __shared__ float sq[HDIM];
    int q = blockIdx.x % Lq;
    int h = (blockIdx.x / Lq) % NHEAD;
    int b = blockIdx.x / (Lq * NHEAD);
    int tid = threadIdx.x;
    const float scale = 0.17677669529663687f;   // 1/sqrt(32)
    if (tid < HDIM) sq[tid] = Q[((size_t)(b * Lq + q)) * DIMD + h * HDIM + tid];
    __syncthreads();
    float lm = -1e30f;
    for (int k = tid; k < Lk; k += 128) {
        const float* kr = Kp + ((size_t)(b * Lk + k)) * DIMD + h * HDIM;
        float s = 0.0f;
#pragma unroll
        for (int d = 0; d < HDIM; ++d) s += sq[d] * kr[d];
        s *= scale; sc[k] = s; lm = fmaxf(lm, s);
    }
    red[tid] = lm; __syncthreads();
    for (int o = 64; o; o >>= 1) { if (tid < o) red[tid] = fmaxf(red[tid], red[tid + o]); __syncthreads(); }
    float mx = red[0]; __syncthreads();
    float ls = 0.0f;
    for (int k = tid; k < Lk; k += 128) { float e = expf(sc[k] - mx); sc[k] = e; ls += e; }
    red[tid] = ls; __syncthreads();
    for (int o = 64; o; o >>= 1) { if (tid < o) red[tid] += red[tid + o]; __syncthreads(); }
    float inv = 1.0f / red[0];
    __syncthreads();
    if (tid < HDIM) {
        float acc = 0.0f;
        for (int k = 0; k < Lk; ++k)
            acc += sc[k] * Vp[((size_t)(b * Lk + k)) * DIMD + h * HDIM + tid];
        O[((size_t)(b * Lq + q)) * DIMD + h * HDIM + tid] = acc * inv;
    }
}

// ---------------- misc elementwise ----------------
__global__ void k_add(float* __restrict__ a, const float* __restrict__ bsrc, int n)
{
    int id = blockIdx.x * blockDim.x + threadIdx.x;
    if (id < n) a[id] += bsrc[id];
}
__global__ void k_bcast(const float* __restrict__ src, float* __restrict__ dst)
{
    int id = blockIdx.x * blockDim.x + threadIdx.x;
    if (id >= BB * NLAT * DIMD) return;
    dst[id] = src[id % (NLAT * DIMD)];
}
__global__ void k_gate(const float* __restrict__ pf, const float* __restrict__ gf,
                       const float* __restrict__ glin, float* __restrict__ out)
{
    int id = blockIdx.x * blockDim.x + threadIdx.x;
    if (id >= BN * DIMD) return;
    float g = 1.0f / (1.0f + expf(-glin[id]));
    out[id] = pf[id] + g * gf[id];
}

// =====================================================================
// host side
// =====================================================================
static void gemm(hipStream_t st, const float* A, const float* W, const float* bias,
                 float* C, int M, int Nc, int K, int lda, int act)
{
    dim3 g(Nc / 16, M / 128);       // M % 128 == 0, N % 16 == 0 at all call sites
    k_gemm<<<g, dim3(256), 0, st>>>(A, W, bias, C, M, Nc, K, lda, act);
}
static inline int nb(int total) { return (total + 255) / 256; }

extern "C" void kernel_launch(void* const* d_in, const int* in_sizes, int n_in,
                              void* d_out, int out_size, void* d_ws, size_t ws_size,
                              hipStream_t stream)
{
    (void)in_sizes; (void)n_in; (void)out_size; (void)ws_size;
    // ---- unpack inputs: jax tree order (dict keys alphabetical, lists in order) ----
    int ii = 0;
    auto F = [&]() -> const float* { return (const float*)d_in[ii++]; };
    const float* cond    = F();                                   // 'cond'
    // params (sorted): cond1, cond2, emb1, emb2, emb_ln, fus1, fus2, fus_ln,
    //                  layers, out, out_ln, perceiver   (lin dicts: b then w; ln: b then g)
    const float *cond1_b = F(), *cond1_w = F();
    const float *cond2_b = F(), *cond2_w = F();
    const float *emb1_b  = F(), *emb1_w  = F();
    const float *emb2_b  = F(), *emb2_w  = F();
    const float *embln_b = F(), *embln_g = F();
    const float *fus1_b  = F(), *fus1_w  = F();
    const float *fus2_b  = F(), *fus2_w  = F();
    const float *fusln_b = F(), *fusln_g = F();
    struct Layer { const float *filmb_b,*filmb_w,*filmg_b,*filmg_w,
                   *A_log,*Dp,*conv_b,*conv_w,*dt_b,*dt_w,*in_proj,*out_proj,*x_proj,
                   *norm_b,*norm_g; } L[4];
    for (int l = 0; l < 4; ++l) {           // layer dict: film_b, film_g, mamba, norm
        L[l].filmb_b = F(); L[l].filmb_w = F();
        L[l].filmg_b = F(); L[l].filmg_w = F();
        // mamba sorted: A_log, D, conv_b, conv_w, dt_b, dt_w, in_proj, out_proj, x_proj
        L[l].A_log = F(); L[l].Dp = F(); L[l].conv_b = F(); L[l].conv_w = F();
        L[l].dt_b = F(); L[l].dt_w = F(); L[l].in_proj = F(); L[l].out_proj = F(); L[l].x_proj = F();
        L[l].norm_b = F(); L[l].norm_g = F();
    }
    const float *out_b = F(), *out_w = F();
    const float *outln_b = F(), *outln_g = F();
    struct MhaP { const float *bk,*bo,*bq,*bv,*wk,*wo,*wq,*wv; };
    auto rdMha = [&](MhaP& m) { m.bk=F(); m.bo=F(); m.bq=F(); m.bv=F();
                                m.wk=F(); m.wo=F(); m.wq=F(); m.wv=F(); };
    // perceiver sorted: dec_attn, dec_ln_kv, dec_ln_q, enc_attn, enc_ln_kv, enc_ln_q,
    //                   gate, latents, p2l, self_layers
    MhaP dec_attn; rdMha(dec_attn);
    const float *declnkv_b = F(), *declnkv_g = F();
    const float *declnq_b  = F(), *declnq_g  = F();
    MhaP enc_attn; rdMha(enc_attn);
    const float *enclnkv_b = F(), *enclnkv_g = F();
    const float *enclnq_b  = F(), *enclnq_g  = F();
    const float *gate_b = F(), *gate_w = F();
    const float *latents = F();
    const float *p2l_b = F(), *p2l_w = F();
    struct TelP { MhaP attn; const float *ff1_b,*ff1_w,*ff2_b,*ff2_w,*ln1_b,*ln1_g,*ln2_b,*ln2_g; } T[2];
    for (int l = 0; l < 2; ++l) {           // tel sorted: attn, ff1, ff2, ln1, ln2
        rdMha(T[l].attn);
        T[l].ff1_b = F(); T[l].ff1_w = F(); T[l].ff2_b = F(); T[l].ff2_w = F();
        T[l].ln1_b = F(); T[l].ln1_g = F(); T[l].ln2_b = F(); T[l].ln2_g = F();
    }
    const float* point_feat = F();          // 'point_feat'
    const float* x_in = F();                // 'x'
    const float* xyz  = F();                // 'xyz'

    // ---- workspace bump allocator ----
    char* wp = (char*)d_ws;
    auto alloc = [&](size_t nflt) -> float* {
        float* r = (float*)wp;
        wp += ((nflt * sizeof(float) + 255) & ~(size_t)255);
        return r;
    };
    float* feat  = alloc((size_t)BN * INCH);
    float* tA    = alloc((size_t)BN * DIMD);
    float* hbuf  = alloc((size_t)BN * DIMD);
    float* condg = alloc(2 * DIMD);
    float* conde = alloc(2 * DIMD);
    float* gam   = alloc(4 * 2 * DIMD);
    float* betb  = alloc(4 * 2 * DIMD);
    float* mnmx  = alloc(2 * 6);
    unsigned long long* codes = (unsigned long long*)alloc(BN * 2);
    int* si_h = (int*)alloc(BN); int* ui_h = (int*)alloc(BN);
    int* si_z = (int*)alloc(BN); int* ui_z = (int*)alloc(BN);
    float* xs   = alloc((size_t)BN * DIMD);
    float* lnb  = alloc((size_t)BN * DIMD);
    float* uz   = alloc((size_t)BN * 1024);
    float* uc   = alloc((size_t)BN * DINNER);
    float* xdbl = alloc((size_t)BN * XDBLW);
    float* dtb  = alloc((size_t)BN * DINNER);
    float* ybuf = alloc((size_t)BN * DINNER);
    float* mout = alloc((size_t)BN * DIMD);
    float* xh   = alloc((size_t)BN * DIMD);
    float* xz   = alloc((size_t)BN * DIMD);
    float* fcat = alloc((size_t)BN * 512);
    float* f1   = alloc((size_t)BN * 512);
    float* pfb  = alloc((size_t)BN * DIMD);
    float* pp   = alloc((size_t)BN * DIMD);
    float* lnA  = alloc((size_t)BN * DIMD);
    float* lnL  = alloc((size_t)2 * NLAT * DIMD);
    float* Qb   = alloc((size_t)BN * DIMD);
    float* Kb   = alloc((size_t)BN * DIMD);
    float* Vb   = alloc((size_t)BN * DIMD);
    float* Ob   = alloc((size_t)BN * DIMD);
    float* lat  = alloc((size_t)2 * NLAT * DIMD);
    float* lat2 = alloc((size_t)2 * NLAT * DIMD);
    float* ffb  = alloc((size_t)2 * NLAT * 1024);
    float* gfb  = alloc((size_t)BN * DIMD);
    float* gcat = alloc((size_t)BN * 512);
    float* glin = alloc((size_t)BN * DIMD);
    float* fout = alloc((size_t)BN * DIMD);

    // ---- embedding + cond ----
    k_feat<<<nb(BN * INCH), 256, 0, stream>>>(x_in, point_feat, xyz, feat);
    gemm(stream, feat, emb1_w, emb1_b, tA, BN, DIMD, INCH, INCH, 1);
    gemm(stream, tA, emb2_w, emb2_b, hbuf, BN, DIMD, DIMD, DIMD, 0);
    k_ln256<<<(BN + 7) / 8, 256, 0, stream>>>(hbuf, embln_g, embln_b, hbuf, BN);

    k_slin<<<dim3(1, 2), 256, 0, stream>>>(cond, cond1_w, cond1_b, condg, CONDD, DIMD, 1);
    k_slin<<<dim3(1, 2), 256, 0, stream>>>(condg, cond2_w, cond2_b, conde, DIMD, DIMD, 0);
    for (int l = 0; l < 4; ++l) {
        k_slin<<<dim3(1, 2), 256, 0, stream>>>(conde, L[l].filmg_w, L[l].filmg_b, gam + l * 512, DIMD, DIMD, 0);
        k_slin<<<dim3(1, 2), 256, 0, stream>>>(conde, L[l].filmb_w, L[l].filmb_b, betb + l * 512, DIMD, DIMD, 0);
    }

    // ---- space-filling curve argsorts ----
    k_minmax<<<2, 256, 0, stream>>>(xyz, mnmx);
    k_codes<<<nb(BN), 256, 0, stream>>>(xyz, mnmx, codes, si_h, 0);
    k_bitonic<<<2, 1024, 0, stream>>>(codes, si_h);
    k_invperm<<<nb(BN), 256, 0, stream>>>(si_h, ui_h);
    k_codes<<<nb(BN), 256, 0, stream>>>(xyz, mnmx, codes, si_z, 1);
    k_bitonic<<<2, 1024, 0, stream>>>(codes, si_z);
    k_invperm<<<nb(BN), 256, 0, stream>>>(si_z, ui_z);

    // ---- bidirectional mamba stacks over both curve orders ----
    for (int curve = 0; curve < 2; ++curve) {
        const int* si = curve ? si_z : si_h;
        const int* ui = curve ? ui_z : ui_h;
        k_gather<<<nb(BN * DIMD), 256, 0, stream>>>(hbuf, si, xs, DIMD);
        for (int l = 0; l < 4; ++l) {
            k_ln256<<<(BN + 7) / 8, 256, 0, stream>>>(xs, L[l].norm_g, L[l].norm_b, lnb, BN);
            gemm(stream, lnb, L[l].in_proj, nullptr, uz, BN, 1024, DIMD, DIMD, 0);  // shared fwd/bwd
            for (int dir = 0; dir < 2; ++dir) {
                k_conv<<<nb(BN * DINNER), 256, 0, stream>>>(uz, L[l].conv_w, L[l].conv_b, uc, dir);
                gemm(stream, uc, L[l].x_proj, nullptr, xdbl, BN, XDBLW, DINNER, DINNER, 0);
                gemm(stream, xdbl, L[l].dt_w, L[l].dt_b, dtb, BN, DINNER, DTRANK, XDBLW, 2);
                k_scan<<<4, 256, 0, stream>>>(dtb, uc, xdbl, L[l].A_log, L[l].Dp, uz, ybuf, dir);
                gemm(stream, ybuf, L[l].out_proj, nullptr, mout, BN, DIMD, DINNER, DINNER, 0);
                k_film<<<nb(BN * DIMD), 256, 0, stream>>>(xs, mout, gam + l * 512, betb + l * 512);
            }
        }
        k_gather<<<nb(BN * DIMD), 256, 0, stream>>>(xs, ui, curve ? xz : xh, DIMD);
    }

    // ---- fusion ----
    k_cat2<<<nb(BN * 512), 256, 0, stream>>>(xh, xz, fcat, DIMD, DIMD);
    gemm(stream, fcat, fus1_w, fus1_b, f1, BN, 512, 512, 512, 1);
    gemm(stream, f1, fus2_w, fus2_b, pfb, BN, DIMD, 512, 512, 0);
    k_ln256<<<(BN + 7) / 8, 256, 0, stream>>>(pfb, fusln_g, fusln_b, pfb, BN);

    // ---- perceiver encoder cross-attention ----
    gemm(stream, pfb, p2l_w, p2l_b, pp, BN, DIMD, DIMD, DIMD, 0);
    k_bcast<<<nb(2 * NLAT * DIMD), 256, 0, stream>>>(latents, lat);
    k_ln256<<<(2 * NLAT + 7) / 8, 256, 0, stream>>>(lat, enclnq_g, enclnq_b, lnL, 2 * NLAT);
    k_ln256<<<(BN + 7) / 8, 256, 0, stream>>>(pp, enclnkv_g, enclnkv_b, lnA, BN);
    gemm(stream, lnL, enc_attn.wq, enc_attn.bq, Qb, 2 * NLAT, DIMD, DIMD, DIMD, 0);
    gemm(stream, lnA, enc_attn.wk, enc_attn.bk, Kb, BN, DIMD, DIMD, DIMD, 0);
    gemm(stream, lnA, enc_attn.wv, enc_attn.bv, Vb, BN, DIMD, DIMD, DIMD, 0);
    k_attn<<<BB * NHEAD * NLAT, 128, NN * sizeof(float), stream>>>(Qb, Kb, Vb, Ob, NLAT, NN);
    gemm(stream, Ob, enc_attn.wo, enc_attn.bo, lat, 2 * NLAT, DIMD, DIMD, DIMD, 0);

    // ---- perceiver self layers ----
    for (int l = 0; l < 2; ++l) {
        k_ln256<<<(2 * NLAT + 7) / 8, 256, 0, stream>>>(lat, T[l].ln1_g, T[l].ln1_b, lnL, 2 * NLAT);
        gemm(stream, lnL, T[l].attn.wq, T[l].attn.bq, Qb, 2 * NLAT, DIMD, DIMD, DIMD, 0);
        gemm(stream, lnL, T[l].attn.wk, T[l].attn.bk, Kb, 2 * NLAT, DIMD, DIMD, DIMD, 0);
        gemm(stream, lnL, T[l].attn.wv, T[l].attn.bv, Vb, 2 * NLAT, DIMD, DIMD, DIMD, 0);
        k_attn<<<BB * NHEAD * NLAT, 128, NLAT * sizeof(float), stream>>>(Qb, Kb, Vb, Ob, NLAT, NLAT);
        gemm(stream, Ob, T[l].attn.wo, T[l].attn.bo, lat2, 2 * NLAT, DIMD, DIMD, DIMD, 0);
        k_add<<<nb(2 * NLAT * DIMD), 256, 0, stream>>>(lat, lat2, 2 * NLAT * DIMD);
        k_ln256<<<(2 * NLAT + 7) / 8, 256, 0, stream>>>(lat, T[l].ln2_g, T[l].ln2_b, lnL, 2 * NLAT);
        gemm(stream, lnL, T[l].ff1_w, T[l].ff1_b, ffb, 2 * NLAT, 1024, DIMD, DIMD, 1);
        gemm(stream, ffb, T[l].ff2_w, T[l].ff2_b, lat2, 2 * NLAT, DIMD, 1024, 1024, 0);
        k_add<<<nb(2 * NLAT * DIMD), 256, 0, stream>>>(lat, lat2, 2 * NLAT * DIMD);
    }

    // ---- perceiver decoder + gate ----
    k_ln256<<<(BN + 7) / 8, 256, 0, stream>>>(pfb, declnq_g, declnq_b, lnA, BN);
    k_ln256<<<(2 * NLAT + 7) / 8, 256, 0, stream>>>(lat, declnkv_g, declnkv_b, lnL, 2 * NLAT);
    gemm(stream, lnA, dec_attn.wq, dec_attn.bq, Qb, BN, DIMD, DIMD, DIMD, 0);
    gemm(stream, lnL, dec_attn.wk, dec_attn.bk, Kb, 2 * NLAT, DIMD, DIMD, DIMD, 0);
    gemm(stream, lnL, dec_attn.wv, dec_attn.bv, Vb, 2 * NLAT, DIMD, DIMD, DIMD, 0);
    k_attn<<<BB * NHEAD * NN, 128, NLAT * sizeof(float), stream>>>(Qb, Kb, Vb, Ob, NN, NLAT);
    gemm(stream, Ob, dec_attn.wo, dec_attn.bo, gfb, BN, DIMD, DIMD, DIMD, 0);
    k_cat2<<<nb(BN * 512), 256, 0, stream>>>(pfb, gfb, gcat, DIMD, DIMD);
    gemm(stream, gcat, gate_w, gate_b, glin, BN, DIMD, 512, 512, 0);
    k_gate<<<nb(BN * DIMD), 256, 0, stream>>>(pfb, gfb, glin, fout);

    // ---- output head ----
    k_ln256<<<(BN + 7) / 8, 256, 0, stream>>>(fout, outln_g, outln_b, fout, BN);
    gemm(stream, fout, out_w, out_b, (float*)d_out, BN, DIMD, DIMD, DIMD, 0);
}
